// Seq2SeqDec_19078244729065
// MI455X (gfx1250) — matrix operand order repaired and checked by reading, not compiled, not run on hardware
//
#include <hip/hip_runtime.h>
#include <stdint.h>

typedef __bf16 bf16;
typedef __attribute__((ext_vector_type(16))) __bf16 v16bf;
typedef __attribute__((ext_vector_type(8)))  __bf16 v8bf;
typedef __attribute__((ext_vector_type(8)))  float  v8f;
typedef __attribute__((ext_vector_type(4)))  uint32_t u32x4;
typedef __attribute__((ext_vector_type(8)))  uint32_t u32x8;

#define B_   8
#define T_   10
#define C_   64
#define HW_  4096
#define NT_  3
#define K1_  1152   // (64+64)*9
#define K3_  576    // 64*9

__device__ __forceinline__ float hsig(float x){ return fminf(fmaxf(0.2f*x + 0.5f, 0.0f), 1.0f); }
__device__ __forceinline__ float sigm(float x){ return 1.0f / (1.0f + __expf(-x)); }
__device__ __forceinline__ int   clamp63(int v){ return v < 0 ? 0 : (v > 63 ? 63 : v); }

// ---------------------------------------------------------------------------
// Tensor Data Mover: 2D tile (rows x 64B) from global into LDS, with LDS
// padding 64B->+16B so rows land at an 80-byte pitch (bank-conflict friendly,
// matches the WMMA fragment layout below).  data_size = 8 bytes.
// Issued once per wave (EXEC ignored by TDM); gate with wv==0.
// ---------------------------------------------------------------------------
__device__ __forceinline__ void tdm_load_2d(uint32_t lds_addr, const void* gaddr,
                                            uint32_t tensor_d0_8B, uint32_t rows,
                                            uint32_t row_stride_8B)
{
    uint64_t ga = (uint64_t)(uintptr_t)gaddr;
    u32x4 g0;
    g0[0] = 1u;                                            // count=1, user descriptor
    g0[1] = lds_addr;                                      // lds_addr (bytes)
    g0[2] = (uint32_t)ga;                                  // global addr [31:0]
    g0[3] = (uint32_t)((ga >> 32) & 0x01ffffffu)           // global addr [56:32]
          | (2u << 30);                                    // type = 2 (image)
    u32x8 g1;
    g1[0] = (3u << 16)                                     // data_size = 8B
          | (1u << 20)                                     // pad_enable
          | (3u << 22)                                     // pad_interval: 16 DWORDs (64B)
          | (3u << 25);                                    // pad_amount:   4 DWORDs (16B)
    g1[1] = (tensor_d0_8B & 0xffffu) << 16;                // tensor_dim0 [15:0]
    g1[2] = (tensor_d0_8B >> 16) | ((rows & 0xffffu) << 16); // td0 hi, tensor_dim1 lo
    g1[3] = (rows >> 16) | (8u << 16);                     // td1 hi, tile_dim0 = 8 (64B)
    g1[4] = rows & 0xffffu;                                // tile_dim1 = rows, tile_dim2=0
    g1[5] = row_stride_8B;                                 // tensor_dim0_stride [31:0]
    g1[6] = 0u;                                            // stride hi, dim1_stride (2D)
    g1[7] = 0u;
    asm volatile("tensor_load_to_lds %0, %1" :: "s"(g0), "s"(g1) : "memory");
}

// ---------------------------------------------------------------------------
// Fused ConvLSTM step: implicit GEMM  M=64 (one image row) x N=256 (all 4
// gates) x K=1152 ([x|h] 3x3).  Weights streamed by TDM (double-buffered),
// epilogue does bias + gates + cell update (+ fused BatchNorm for layer 1).
// ---------------------------------------------------------------------------
__global__ __launch_bounds__(256)
void convlstm_step(const bf16* __restrict__ xin,   // (B,T,64,H,W) bf16 sequence
                   const bf16* __restrict__ hin,   // (B,64,H,W)
                   float*      __restrict__ cst,   // (B,64,H,W) in/out
                   bf16*       __restrict__ hout,  // (B,64,H,W)
                   bf16*       __restrict__ seq,   // (B,T,64,H,W) layer output
                   const bf16* __restrict__ Wcat,  // (256,1152) = [Wx|Wh] bf16
                   const float* __restrict__ bias, // (256)
                   int t, int apply_bn,
                   const float* __restrict__ gam, const float* __restrict__ bet,
                   const float* __restrict__ mu,  const float* __restrict__ var)
{
    __shared__ alignas(16) bf16 Asm[64 * 40];        // 64 pixels x 32 K (pitch 40)
    __shared__ alignas(16) bf16 Bsm[2][256 * 40];    // double-buffered weight tile

    const int tid  = threadIdx.x;
    const int lane = tid & 31;
    const int wv   = tid >> 5;       // 8 waves
    const int blk  = blockIdx.x;     // B*H = 512
    const int b    = blk >> 6;
    const int y    = blk & 63;

    const int mi = wv >> 1;          // M-tile 0..3
    const int qb = (wv & 1) * 2;     // q base 0 or 2

    v8f acc[4][2];
    #pragma unroll
    for (int g = 0; g < 4; ++g)
        #pragma unroll
        for (int p = 0; p < 2; ++p) acc[g][p] = v8f{};

    // A-gather mapping: fixed k per thread, 8 consecutive x values
    const int klocal = tid >> 3;         // 0..31
    const int mseg   = (tid & 7) * 8;    // x base

    const int   frame = b * T_ + t;
    const bf16* xf = xin + (long)frame * (C_ * HW_);
    const bf16* hb = hin + (long)b * (C_ * HW_);

    const uint32_t bbase0 = (uint32_t)(uintptr_t)&Bsm[0][0];
    const uint32_t bbase1 = (uint32_t)(uintptr_t)&Bsm[1][0];

    if (wv == 0) tdm_load_2d(bbase0, Wcat, K1_ / 4, 256, K1_ / 4);  // 288 x 8B rows
    int cur = 0;

    const int NITER = K1_ / 32;   // 36
    for (int kk = 0; kk < NITER; ++kk) {
        const int k0 = kk * 32;
        // ---- A tile: im2row (x: symmetric pad -> clamp; h: zero pad)
        {
            int k  = k0 + klocal;
            int c  = k / 9;
            int r  = k - c * 9;
            int ky = r / 3;
            int kx = r - ky * 3;
            if (c < 64) {
                int yy = clamp63(y + ky - 1);
                const bf16* src = xf + c * HW_ + yy * 64;
                #pragma unroll
                for (int j = 0; j < 8; ++j)
                    Asm[(mseg + j) * 40 + klocal] = src[clamp63(mseg + j + kx - 1)];
            } else {
                int yy = y + ky - 1;
                if ((unsigned)yy < 64u) {
                    const bf16* src = hb + (c - 64) * HW_ + yy * 64;
                    #pragma unroll
                    for (int j = 0; j < 8; ++j) {
                        int xx = mseg + j + kx - 1;
                        Asm[(mseg + j) * 40 + klocal] =
                            ((unsigned)xx < 64u) ? src[xx] : (bf16)0.0f;
                    }
                } else {
                    #pragma unroll
                    for (int j = 0; j < 8; ++j)
                        Asm[(mseg + j) * 40 + klocal] = (bf16)0.0f;
                }
            }
        }
        // ---- B tile: TDM prefetch next, wait for current
        if (wv == 0) {
            if (kk + 1 < NITER) {
                tdm_load_2d(cur ? bbase0 : bbase1, Wcat + (kk + 1) * 32,
                            K1_ / 4, 256, K1_ / 4);
                __builtin_amdgcn_s_wait_tensorcnt(1);
            } else {
                __builtin_amdgcn_s_wait_tensorcnt(0);
            }
        }
        __syncthreads();

        const bf16* Bt = &Bsm[cur][0];
        // ---- A fragment (16x32 bf16 per ISA layout)
        const int mrow = mi * 16 + (lane & 15);
        const int ka   = (lane < 16) ? 0 : 8;
        v16bf af;
        {
            v8bf lo = *(const v8bf*)(Asm + mrow * 40 + ka);
            v8bf hi = *(const v8bf*)(Asm + mrow * 40 + ka + 16);
            #pragma unroll
            for (int i = 0; i < 8; ++i) { af[i] = lo[i]; af[i + 8] = hi[i]; }
        }
        // ---- B fragments + WMMA
        const int kb   = (lane < 16) ? 0 : 8;
        const int ncol = lane & 15;
        #pragma unroll
        for (int g = 0; g < 4; ++g) {
            #pragma unroll
            for (int p = 0; p < 2; ++p) {
                const int nt = g * 4 + qb + p;
                const bf16* bp = Bt + (nt * 16 + ncol) * 40 + kb;
                v8bf blo = *(const v8bf*)bp;
                v8bf bhi = *(const v8bf*)(bp + 16);
                v16bf bm;
                #pragma unroll
                for (int i = 0; i < 8; ++i) { bm[i] = blo[i]; bm[i + 8] = bhi[i]; }
                acc[g][p] = __builtin_amdgcn_wmma_f32_16x16x32_bf16(
                    false, af, false, bm, (short)0, acc[g][p], false, false);
            }
        }
        __syncthreads();
        cur ^= 1;
    }

    // ---- Epilogue: i,f,c,o for same (pixel, channel) are local to this lane
    const int e8  = (lane >> 4) * 8;
    const int nlo = lane & 15;
    float bns[8], bnb[8];
    if (apply_bn) {
        #pragma unroll
        for (int e = 0; e < 8; ++e) {
            int x = mi * 16 + e + e8;
            float s = gam[x] * rsqrtf(var[x] + 1e-3f);
            bns[e] = s;
            bnb[e] = bet[x] - mu[x] * s;
        }
    }
    #pragma unroll
    for (int p = 0; p < 2; ++p) {
        const int  cidx = (qb + p) * 16 + nlo;
        const float bi = bias[cidx], bf_ = bias[64 + cidx];
        const float bc = bias[128 + cidx], bo = bias[192 + cidx];
        const long cbase = ((long)b * C_ + cidx) * HW_ + y * 64;
        const long sbase = ((long)frame * C_ + cidx) * HW_ + y * 64;
        #pragma unroll
        for (int e = 0; e < 8; ++e) {
            int x = mi * 16 + e + e8;
            float iv = hsig(acc[0][p][e] + bi);
            float fv = hsig(acc[1][p][e] + bf_);
            float gv = sigm(acc[2][p][e] + bc);
            float ov = hsig(acc[3][p][e] + bo);
            float cn = fv * cst[cbase + x] + iv * gv;
            cst[cbase + x] = cn;
            float hv = ov * sigm(cn);
            hout[cbase + x] = (bf16)hv;
            float sv = apply_bn ? (hv * bns[e] + bnb[e]) : hv;
            seq[sbase + x] = (bf16)sv;
        }
    }
}

// ---------------------------------------------------------------------------
// F->F 3x3 conv + ReLU (symmetric pad), implicit GEMM M=64 x N=64 x K=576
// ---------------------------------------------------------------------------
__global__ __launch_bounds__(256)
void conv_relu64(const bf16* __restrict__ xin,   // (frames,64,HW)
                 bf16*       __restrict__ out,   // (frames,64,HW)
                 const bf16* __restrict__ Wk,    // (64,576)
                 const float* __restrict__ bias) // (64)
{
    __shared__ alignas(16) bf16 Asm[64 * 40];
    __shared__ alignas(16) bf16 Bsm[2][64 * 40];

    const int tid  = threadIdx.x;
    const int lane = tid & 31;
    const int wv   = tid >> 5;
    const int blk  = blockIdx.x;      // frames*H = 5120
    const int f    = blk >> 6;
    const int y    = blk & 63;

    const int mi = wv >> 1;
    const int nb = (wv & 1) * 2;

    v8f acc[2];
    acc[0] = v8f{}; acc[1] = v8f{};

    const int klocal = tid >> 3;
    const int mseg   = (tid & 7) * 8;
    const bf16* xf = xin + (long)f * (C_ * HW_);

    const uint32_t bbase0 = (uint32_t)(uintptr_t)&Bsm[0][0];
    const uint32_t bbase1 = (uint32_t)(uintptr_t)&Bsm[1][0];
    if (wv == 0) tdm_load_2d(bbase0, Wk, K3_ / 4, 64, K3_ / 4);  // 144 x 8B rows
    int cur = 0;

    const int NITER = K3_ / 32;   // 18
    for (int kk = 0; kk < NITER; ++kk) {
        const int k0 = kk * 32;
        {
            int k  = k0 + klocal;
            int c  = k / 9;
            int r  = k - c * 9;
            int ky = r / 3;
            int kx = r - ky * 3;
            int yy = clamp63(y + ky - 1);
            const bf16* src = xf + c * HW_ + yy * 64;
            #pragma unroll
            for (int j = 0; j < 8; ++j)
                Asm[(mseg + j) * 40 + klocal] = src[clamp63(mseg + j + kx - 1)];
        }
        if (wv == 0) {
            if (kk + 1 < NITER) {
                tdm_load_2d(cur ? bbase0 : bbase1, Wk + (kk + 1) * 32,
                            K3_ / 4, 64, K3_ / 4);
                __builtin_amdgcn_s_wait_tensorcnt(1);
            } else {
                __builtin_amdgcn_s_wait_tensorcnt(0);
            }
        }
        __syncthreads();

        const bf16* Bt = &Bsm[cur][0];
        const int mrow = mi * 16 + (lane & 15);
        const int ka   = (lane < 16) ? 0 : 8;
        v16bf af;
        {
            v8bf lo = *(const v8bf*)(Asm + mrow * 40 + ka);
            v8bf hi = *(const v8bf*)(Asm + mrow * 40 + ka + 16);
            #pragma unroll
            for (int i = 0; i < 8; ++i) { af[i] = lo[i]; af[i + 8] = hi[i]; }
        }
        const int kb   = (lane < 16) ? 0 : 8;
        const int ncol = lane & 15;
        #pragma unroll
        for (int p = 0; p < 2; ++p) {
            const int nt = nb + p;
            const bf16* bp = Bt + (nt * 16 + ncol) * 40 + kb;
            v8bf blo = *(const v8bf*)bp;
            v8bf bhi = *(const v8bf*)(bp + 16);
            v16bf bm;
            #pragma unroll
            for (int i = 0; i < 8; ++i) { bm[i] = blo[i]; bm[i + 8] = bhi[i]; }
            acc[p] = __builtin_amdgcn_wmma_f32_16x16x32_bf16(
                false, af, false, bm, (short)0, acc[p], false, false);
        }
        __syncthreads();
        cur ^= 1;
    }

    const int e8  = (lane >> 4) * 8;
    const int nlo = lane & 15;
    #pragma unroll
    for (int p = 0; p < 2; ++p) {
        const int n  = (nb + p) * 16 + nlo;
        const float bv = bias[n];
        const long obase = ((long)f * C_ + n) * HW_ + y * 64;
        #pragma unroll
        for (int e = 0; e < 8; ++e) {
            int x = mi * 16 + e + e8;
            out[obase + x] = (bf16)fmaxf(acc[p][e] + bv, 0.0f);
        }
    }
}

// ---------------------------------------------------------------------------
// F->NT (3) 3x3 conv + ReLU, direct (tiny: ~1 GFLOP), f32 output
// ---------------------------------------------------------------------------
__global__ __launch_bounds__(256)
void conv_out3(const bf16* __restrict__ xin,   // (frames,64,HW)
               float*      __restrict__ out,   // (frames,3,HW)
               const bf16* __restrict__ Wk,    // (3,576)
               const float* __restrict__ bias) // (3)
{
    __shared__ bf16 Ws[NT_ * K3_];
    for (int i = threadIdx.x; i < NT_ * K3_; i += 256) Ws[i] = Wk[i];
    __syncthreads();

    int idx = blockIdx.x * 256 + threadIdx.x;        // frames*3*4096
    if (idx >= 80 * NT_ * HW_) return;
    int f  = idx / (NT_ * HW_);
    int r  = idx - f * (NT_ * HW_);
    int o  = r >> 12;
    int yx = r & 4095;
    int y  = yx >> 6;
    int x  = yx & 63;

    const bf16* xf = xin + (long)f * (C_ * HW_);
    float a = bias[o];
    for (int c = 0; c < 64; ++c) {
        #pragma unroll
        for (int ky = 0; ky < 3; ++ky) {
            int yy = clamp63(y + ky - 1);
            #pragma unroll
            for (int kx = 0; kx < 3; ++kx) {
                int xx = clamp63(x + kx - 1);
                a += (float)xf[c * HW_ + yy * 64 + xx] *
                     (float)Ws[o * K3_ + c * 9 + ky * 3 + kx];
            }
        }
    }
    out[idx] = fmaxf(a, 0.0f);
}

// ---------------------------------------------------------------------------
// Prep kernels
// ---------------------------------------------------------------------------
__global__ void cvt_bf16(const float* __restrict__ src, bf16* __restrict__ dst, int n)
{
    int i = blockIdx.x * 256 + threadIdx.x;
    if (i < n) dst[i] = (bf16)src[i];
}

__global__ void pack_gates(const float* __restrict__ wx, const float* __restrict__ wh,
                           bf16* __restrict__ dst)   // (256, 1152)
{
    int i = blockIdx.x * 256 + threadIdx.x;
    if (i >= 256 * K1_) return;
    int r = i / K1_, k = i - r * K1_;
    float v = (k < K3_) ? wx[r * K3_ + k] : wh[r * K3_ + (k - K3_)];
    dst[i] = (bf16)v;
}

// ---------------------------------------------------------------------------
extern "C" void kernel_launch(void* const* d_in, const int* in_sizes, int n_in,
                              void* d_out, int out_size, void* d_ws, size_t ws_size,
                              hipStream_t stream)
{
    (void)in_sizes; (void)n_in; (void)out_size; (void)ws_size;
    const float* inputs = (const float*)d_in[0];
    const float* h1     = (const float*)d_in[1];
    const float* c1     = (const float*)d_in[2];
    const float* h2     = (const float*)d_in[3];
    const float* c2     = (const float*)d_in[4];
    const float* Wx1    = (const float*)d_in[5];
    const float* Wh1    = (const float*)d_in[6];
    const float* b1     = (const float*)d_in[7];
    const float* Wx2    = (const float*)d_in[8];
    const float* Wh2    = (const float*)d_in[9];
    const float* b2     = (const float*)d_in[10];
    const float* conv_w = (const float*)d_in[11];
    const float* conv_b = (const float*)d_in[12];
    const float* out_w  = (const float*)d_in[13];
    const float* out_b  = (const float*)d_in[14];
    const float* gam    = (const float*)d_in[15];
    const float* bet    = (const float*)d_in[16];
    const float* mu     = (const float*)d_in[17];
    const float* var    = (const float*)d_in[18];
    float* out = (float*)d_out;

    const long SEQ  = 80L * C_ * HW_;   // 20,971,520 elems
    const long IMG  = 8L * C_ * HW_;    //  2,097,152 elems

    char* p = (char*)d_ws;
    auto alloc = [&](size_t bytes) -> char* {
        char* r = p; p += (bytes + 255) & ~(size_t)255; return r;
    };
    bf16*  in_bf = (bf16*)alloc(SEQ * 2);
    bf16*  x1_bf = (bf16*)alloc(SEQ * 2);
    bf16*  x2_bf = (bf16*)alloc(SEQ * 2);
    bf16*  x3_bf = (bf16*)alloc(SEQ * 2);
    bf16*  h1a   = (bf16*)alloc(IMG * 2);
    bf16*  h1b   = (bf16*)alloc(IMG * 2);
    bf16*  h2a   = (bf16*)alloc(IMG * 2);
    bf16*  h2b   = (bf16*)alloc(IMG * 2);
    float* c1s   = (float*)alloc(IMG * 4);
    float* c2s   = (float*)alloc(IMG * 4);
    bf16*  Wc1   = (bf16*)alloc(256L * K1_ * 2);
    bf16*  Wc2   = (bf16*)alloc(256L * K1_ * 2);
    bf16*  W3    = (bf16*)alloc(64L * K3_ * 2);
    bf16*  W4    = (bf16*)alloc((long)NT_ * K3_ * 2);

    // ---- precision conversion / weight packing
    cvt_bf16<<<(int)((SEQ + 255) / 256), 256, 0, stream>>>(inputs, in_bf, (int)SEQ);
    cvt_bf16<<<(int)((IMG + 255) / 256), 256, 0, stream>>>(h1, h1a, (int)IMG);
    cvt_bf16<<<(int)((IMG + 255) / 256), 256, 0, stream>>>(h2, h2a, (int)IMG);
    hipMemcpyAsync(c1s, c1, IMG * 4, hipMemcpyDeviceToDevice, stream);
    hipMemcpyAsync(c2s, c2, IMG * 4, hipMemcpyDeviceToDevice, stream);
    pack_gates<<<(256 * K1_ + 255) / 256, 256, 0, stream>>>(Wx1, Wh1, Wc1);
    pack_gates<<<(256 * K1_ + 255) / 256, 256, 0, stream>>>(Wx2, Wh2, Wc2);
    cvt_bf16<<<(64 * K3_ + 255) / 256, 256, 0, stream>>>(conv_w, W3, 64 * K3_);
    cvt_bf16<<<(NT_ * K3_ + 255) / 256, 256, 0, stream>>>(out_w, W4, NT_ * K3_);

    // ---- ConvLSTM layer 1 (BN fused into epilogue output)
    for (int t = 0; t < T_; ++t) {
        bf16* hin = (t & 1) ? h1b : h1a;
        bf16* ho  = (t & 1) ? h1a : h1b;
        convlstm_step<<<B_ * 64, 256, 0, stream>>>(
            in_bf, hin, c1s, ho, x1_bf, Wc1, b1, t, 1, gam, bet, mu, var);
    }
    // ---- ConvLSTM layer 2
    for (int t = 0; t < T_; ++t) {
        bf16* hin = (t & 1) ? h2b : h2a;
        bf16* ho  = (t & 1) ? h2a : h2b;
        convlstm_step<<<B_ * 64, 256, 0, stream>>>(
            x1_bf, hin, c2s, ho, x2_bf, Wc2, b2, t, 0, gam, bet, mu, var);
    }
    // ---- F->F conv + ReLU (WMMA)
    conv_relu64<<<80 * 64, 256, 0, stream>>>(x2_bf, x3_bf, W3, conv_b);
    // ---- F->3 conv + ReLU (direct, f32 output)
    conv_out3<<<(80 * NT_ * HW_ + 255) / 256, 256, 0, stream>>>(x3_bf, out, W4, out_b);
}
